// LateralEI_5196910428904
// MI455X (gfx1250) — compile-verified, matrix operand order checked by reading, not run on hardware
//
#include <hip/hip_runtime.h>
#include <hip/hip_bf16.h>

typedef __attribute__((ext_vector_type(16))) _Float16 v16h;
typedef __attribute__((ext_vector_type(8)))  float    v8f;

#define NROWS 8192
#define DHDIM 128
#define WAVES_PER_WG 4

// DoG kernel value for one (i,j) pair.
// 1/(2*0.6^2) = 1.38888..  ==  4 * 1/(2*1.2^2) = 4 * 0.347222..
// so exp(-a*d) = e^4 with e = exp(-b*d): one transcendental per pair.
__device__ __forceinline__ float dog_k(float zx, float zy, float zjx, float zjy) {
  float dx = zx - zjx;
  float dy = zy - zjy;
  float d  = __builtin_fmaf(dy, dy, dx * dx);       // always >= 0 (matches max(d,0))
  const float C = -0.34722222f * 1.44269504f;       // -(1/2.88) * log2(e)
  float e  = __builtin_amdgcn_exp2f(d * C);         // v_exp_f32
  float e2 = e * e;
  float e4 = e2 * e2;
  return __builtin_fmaf(0.8f, e4, -e);              // K_E*e^4 - K_I*e
}

// Prep: h [8192,128] f32 (row major) -> hT [128,8192] f16 so that B-matrix
// tiles are contiguous 32B per lane.
__global__ void __launch_bounds__(256)
h_to_f16_transpose(const float* __restrict__ h, _Float16* __restrict__ hT) {
  int idx = blockIdx.x * 256 + threadIdx.x;   // 0 .. 128*8192-1
  int c = idx >> 13;                          // column of h (0..127)
  int n = idx & (NROWS - 1);                  // row of h
  hT[idx] = (_Float16)h[n * DHDIM + c];
}

// Fused: build 16x32 f16 tile of the normalized-later DoG kernel in registers,
// contract against h with v_wmma_f32_16x16x32_f16, track rowsums via a 9th
// WMMA against an all-ones B matrix, normalize at the end.
__global__ void __launch_bounds__(WAVES_PER_WG * 32)
lateral_ei_wmma(const float* __restrict__ z,
                const _Float16* __restrict__ hT,
                float* __restrict__ out) {
  const int lane = threadIdx.x & 31;
  const int wave = threadIdx.x >> 5;
  const int lm   = lane & 15;                 // A: row M = lm ; C/D: N = lm
  const int hi   = lane >> 4;                 // K-half selector for A/B layout
  const int i0   = (blockIdx.x * WAVES_PER_WG + wave) * 16;

  // This lane's fixed z_i (A-matrix row M = lm).
  const float zx = z[2 * (i0 + lm) + 0];
  const float zy = z[2 * (i0 + lm) + 1];

  v8f acc[8];
  v8f rs;
#pragma unroll
  for (int n = 0; n < 8; ++n)
#pragma unroll
    for (int e = 0; e < 8; ++e) acc[n][e] = 0.0f;
#pragma unroll
  for (int e = 0; e < 8; ++e) rs[e] = 0.0f;

  v16h ones;
#pragma unroll
  for (int t = 0; t < 16; ++t) ones[t] = (_Float16)1.0f;

  for (int j0 = 0; j0 < NROWS; j0 += 32) {
    const int kb = j0 + hi * 16;              // this lane's 16-wide K slice

    // Load 16 consecutive z_j pairs (32 floats, 8x b128, L1/L2 resident).
    float zbuf[32];
    const float4* zq = (const float4*)(z + 2 * kb);
#pragma unroll
    for (int q = 0; q < 8; ++q)
      *(float4*)(zbuf + 4 * q) = zq[q];

    // A-matrix tile: lane holds K(i0+lm, kb+t) for t=0..15, packed 2/VGPR.
    v16h a;
#pragma unroll
    for (int t = 0; t < 16; ++t)
      a[t] = (_Float16)dog_k(zx, zy, zbuf[2 * t], zbuf[2 * t + 1]);

    // Rowsum via ones-matmul: every output column of (A x ones) = rowsum(M),
    // landing in the same C-layout as acc[] -> elementwise normalize later.
    rs = __builtin_amdgcn_wmma_f32_16x16x32_f16(
        false, a, false, ones, (short)0, rs, false, false);

    // 8 N-tiles cover DH=128. B slice: 16 contiguous f16 (32B) per lane.
#pragma unroll
    for (int n = 0; n < 8; ++n) {
      v16h b = *(const v16h*)(hT + (size_t)(n * 16 + lm) * NROWS + kb);
      acc[n] = __builtin_amdgcn_wmma_f32_16x16x32_f16(
          false, a, false, b, (short)0, acc[n], false, false);
    }
  }

  // C/D layout: lanes 0-15 -> M = v, N = lm ; lanes 16-31 -> M = v+8, N = lm.
#pragma unroll
  for (int v = 0; v < 8; ++v) {
    const int row = i0 + hi * 8 + v;
    const float s = 0.05f / (rs[v] + 1e-6f);  // GAIN / (rowsum + 1e-6)
#pragma unroll
    for (int n = 0; n < 8; ++n)
      out[row * DHDIM + n * 16 + lm] = acc[n][v] * s;
  }
}

extern "C" void kernel_launch(void* const* d_in, const int* in_sizes, int n_in,
                              void* d_out, int out_size, void* d_ws, size_t ws_size,
                              hipStream_t stream) {
  const float* z = (const float*)d_in[0];   // [8192, 2]   f32
  const float* h = (const float*)d_in[1];   // [8192, 128] f32
  float* out = (float*)d_out;               // [8192, 128] f32
  _Float16* hT = (_Float16*)d_ws;           // [128, 8192] f16 (2 MB scratch)

  hipLaunchKernelGGL(h_to_f16_transpose,
                     dim3((NROWS * DHDIM) / 256), dim3(256), 0, stream, h, hT);

  hipLaunchKernelGGL(lateral_ei_wmma,
                     dim3(NROWS / (16 * WAVES_PER_WG)), dim3(WAVES_PER_WG * 32),
                     0, stream, z, hT, out);
}
